// ParallelSMLSTMCell_47596827574496
// MI455X (gfx1250) — compile-verified, matrix operand order checked by reading, not run on hardware
//
#include <hip/hip_runtime.h>
#include <hip/hip_bf16.h>
#include <math.h>

typedef __attribute__((ext_vector_type(8)))  _Float16 v8h;
typedef __attribute__((ext_vector_type(16))) _Float16 v16h;
typedef __attribute__((ext_vector_type(8)))  float    v8f;

#define B_   4
#define S_   1024
#define IN_  128
#define D_   128
#define ROWS (B_ * S_)

// ---------------- fast math helpers ----------------
__device__ __forceinline__ float fastRcp(float x) { return __builtin_amdgcn_rcpf(x); }
__device__ __forceinline__ float fastSigmoid(float x) {
  return fastRcp(1.0f + __expf(-x));
}

// ---------------- reductions ----------------
__device__ __forceinline__ float warpSum(float v) {
#pragma unroll
  for (int o = 16; o > 0; o >>= 1) v += __shfl_xor(v, o);
  return v;
}

__device__ __forceinline__ float blockSum(float v, float* sh) {
  v = warpSum(v);
  int wid = threadIdx.x >> 5, nw = blockDim.x >> 5;
  if ((threadIdx.x & 31) == 0) sh[wid] = v;
  __syncthreads();
  float r = (threadIdx.x < nw) ? sh[threadIdx.x] : 0.0f;
  r = warpSum(r);
  if (threadIdx.x == 0) sh[0] = r;
  __syncthreads();
  float out = sh[0];
  __syncthreads();
  return out;
}

// ---------------- WMMA helpers ----------------
// 16-bit A-fragment (16x32): lane&15 = row (A) / col (B), lane>>4 selects the
// {8..15,24..31} K-halves. Two contiguous 16B loads per fragment.
__device__ __forceinline__ v16h load_frag(const _Float16* __restrict__ base, int ld, int k0) {
  int lane = threadIdx.x & 31;
  const _Float16* p = base + (size_t)(lane & 15) * ld + k0 + ((lane >> 4) ? 8 : 0);
  v8h lo = *(const v8h*)(p);
  v8h hi = *(const v8h*)(p + 16);
  v16h r;
#pragma unroll
  for (int i = 0; i < 8; ++i) { r[i] = lo[i]; r[i + 8] = hi[i]; }
  return r;
}

__device__ __forceinline__ v8f wmma_f16(v16h a, v16h b, v8f c) {
  return __builtin_amdgcn_wmma_f32_16x16x32_f16(false, a, false, b, (short)0, c, false, false);
}

// ---------------- small utility kernels ----------------
__global__ void k_f2h(const float* __restrict__ in, _Float16* __restrict__ out, int n) {
  int i = blockIdx.x * blockDim.x + threadIdx.x;
  if (i < n) out[i] = (_Float16)in[i];
}

// LayerNorm(x) + i_tilde/f_tilde dots + stabilized gates.  One row per block.
__global__ void k_ln_gates(const float* __restrict__ x,
                           const float* __restrict__ g, const float* __restrict__ bt,
                           const float* __restrict__ wi, const float* __restrict__ bi,
                           const float* __restrict__ wf, const float* __restrict__ bf,
                           const float* __restrict__ m_init,
                           _Float16* __restrict__ xn16, float* __restrict__ mv,
                           float* __restrict__ igv, float* __restrict__ icv,
                           float* __restrict__ fzv) {
  __shared__ float sh[8];
  int row = blockIdx.x, t = threadIdx.x;
  float v = x[(size_t)row * IN_ + t];
  float mu = blockSum(v, sh) * (1.0f / IN_);
  float dv = v - mu;
  float var = blockSum(dv * dv, sh) * (1.0f / IN_);
  float xn = dv * rsqrtf(var + 1e-5f) * g[t] + bt[t];
  xn16[(size_t)row * IN_ + t] = (_Float16)xn;
  float it = blockSum(xn * wi[t], sh) + bi[0];
  float ft = blockSum(xn * wf[t], sh) + bf[0];
  if (t == 0) {
    mv[row] = mu;  // feature-mean == volatility signal
    float mp = fminf(fmaxf(m_init[0], -100.f), 100.f);
    float mt = fmaxf(ft, mp + ft);
    float ig = __expf(fminf(fmaxf(it - mt, -15.f), 15.f));
    igv[row] = ig;
    icv[row] = fminf(fmaxf(ig, 0.f), 1.f);
    fzv[row] = fminf(fmaxf(ft + mp - mt, -15.f), 15.f);  // log f_g
  }
}

// cumulative log-f per batch (sequential, tiny)
__global__ void k_lscan(const float* __restrict__ fzv, float* __restrict__ Lv) {
  if (threadIdx.x != 0) return;
  int b = blockIdx.x;
  float L = 0.f;
  for (int s = 0; s < S_; ++s) { L += fzv[b * S_ + s]; Lv[b * S_ + s] = L; }
}

// ---------------- fused WMMA GEMM: Y = A[4096x128] @ W^T + b, epilogues ----
// mode 0: f32 out; mode 1: sigmoid; mode 2: LN+gelu -> f16 out;
// mode 3: LN+gelu * regime_w[:,pidx] accumulated into outacc.
__global__ void k_gemm(const _Float16* __restrict__ A, const _Float16* __restrict__ W,
                       const float* __restrict__ bias,
                       float* __restrict__ out32, _Float16* __restrict__ out16,
                       const float* __restrict__ lng, const float* __restrict__ lnb,
                       const float* __restrict__ rw, float* __restrict__ outacc,
                       int mode, int pidx) {
  __shared__ float st[16][128];
  int wave = threadIdx.x >> 5, lane = threadIdx.x & 31;
  int row0 = blockIdx.x * 16;
  int col = wave * 16 + (lane & 15);
  int rh = (lane >> 4) ? 8 : 0;
  v8f acc = {};
  const _Float16* Ab = A + (size_t)row0 * 128;
  const _Float16* Wb = W + (size_t)(wave * 16) * 128;
#pragma unroll
  for (int kk = 0; kk < 128; kk += 32) {
    v16h a = load_frag(Ab, 128, kk);
    v16h b = load_frag(Wb, 128, kk);
    acc = wmma_f16(a, b, acc);
  }
  float bb = bias[col];
#pragma unroll
  for (int i = 0; i < 8; ++i) acc[i] += bb;

  if (mode == 0) {
#pragma unroll
    for (int i = 0; i < 8; ++i) out32[(size_t)(row0 + rh + i) * 128 + col] = acc[i];
  } else if (mode == 1) {
#pragma unroll
    for (int i = 0; i < 8; ++i)
      out32[(size_t)(row0 + rh + i) * 128 + col] = fastSigmoid(acc[i]);
  } else {
#pragma unroll
    for (int i = 0; i < 8; ++i) st[rh + i][col] = acc[i];
    __syncthreads();
    // each wave LayerNorms 2 rows of the 16x128 tile
#pragma unroll
    for (int rr = 0; rr < 2; ++rr) {
      int lr = wave * 2 + rr;
      int grow = row0 + lr;
      float s1 = 0.f, s2 = 0.f;
#pragma unroll
      for (int j = 0; j < 4; ++j) {
        float xv = st[lr][lane * 4 + j];
        s1 += xv; s2 += xv * xv;
      }
      s1 = warpSum(s1); s2 = warpSum(s2);
      float mu = s1 * (1.f / 128.f);
      float var = s2 * (1.f / 128.f) - mu * mu;
      float rstd = rsqrtf(var + 1e-5f);
      float wgt = (mode == 3) ? rw[grow * 4 + pidx] : 0.f;
#pragma unroll
      for (int j = 0; j < 4; ++j) {
        int c = lane * 4 + j;
        float xh = (st[lr][c] - mu) * rstd * lng[c] + lnb[c];
        float gl = 0.5f * xh * (1.0f + erff(xh * 0.70710678118f));
        if (mode == 2) {
          out16[(size_t)grow * 128 + c] = (_Float16)gl;
        } else {
          float vv = wgt * gl;
          if (pidx == 0) outacc[(size_t)grow * 128 + c] = vv;
          else           outacc[(size_t)grow * 128 + c] += vv;
        }
      }
    }
  }
}

// regime_l2 (128->4) + softmax, one row per block
__global__ void k_regime(const _Float16* __restrict__ r16,
                         const float* __restrict__ w2, const float* __restrict__ b2,
                         float* __restrict__ rwout) {
  __shared__ float sh[8];
  int row = blockIdx.x, t = threadIdx.x;
  float rv = (float)r16[(size_t)row * 128 + t];
  float d0 = blockSum(rv * w2[0 * 128 + t], sh) + b2[0];
  float d1 = blockSum(rv * w2[1 * 128 + t], sh) + b2[1];
  float d2 = blockSum(rv * w2[2 * 128 + t], sh) + b2[2];
  float d3 = blockSum(rv * w2[3 * 128 + t], sh) + b2[3];
  if (t == 0) {
    float mx = fmaxf(fmaxf(d0, d1), fmaxf(d2, d3));
    float e0 = __expf(d0 - mx), e1 = __expf(d1 - mx), e2 = __expf(d2 - mx), e3 = __expf(d3 - mx);
    float rs = fastRcp(e0 + e1 + e2 + e3);
    rwout[row * 4 + 0] = e0 * rs; rwout[row * 4 + 1] = e1 * rs;
    rwout[row * 4 + 2] = e2 * rs; rwout[row * 4 + 3] = e3 * rs;
  }
}

// normalize k/v, emit f16 operands (v stored transposed [b][d][s] for B-frags)
__global__ void k_prep(const float* __restrict__ q32, const float* __restrict__ k32,
                       const float* __restrict__ v32,
                       _Float16* __restrict__ q16, _Float16* __restrict__ k16,
                       _Float16* __restrict__ ivT) {
  __shared__ float sh[8];
  int row = blockIdx.x, t = threadIdx.x;
  int b = row >> 10, s = row & 1023;
  float qv = q32[(size_t)row * 128 + t];
  float kv = k32[(size_t)row * 128 + t] * 0.08838834764831845f;  // 1/sqrt(128)
  float vv = v32[(size_t)row * 128 + t];
  float kn2 = blockSum(kv * kv, sh);
  float vn2 = blockSum(vv * vv, sh);
  float kn = kv * fastRcp(fmaxf(sqrtf(kn2), 1e-12f)) * 11.313708498984761f;  // *sqrt(128)
  float vn = vv * fastRcp(sqrtf(vn2) + 1e-6f) * 11.313708498984761f;
  q16[(size_t)row * 128 + t] = (_Float16)qv;
  k16[(size_t)row * 128 + t] = (_Float16)kn;
  ivT[((size_t)(b * 128 + t)) * 1024 + s] = (_Float16)vn;
}

// ---------------- chunked causal linear attention (the mLSTM scan) --------
// grid = B*8, block = 256 (8 waves); wave w owns 16 query rows x 128 d.
__global__ void k_attn(const _Float16* __restrict__ q16, const _Float16* __restrict__ k16,
                       const _Float16* __restrict__ ivT, const float* __restrict__ Lv,
                       const float* __restrict__ igv, const float* __restrict__ icv,
                       float* __restrict__ h32) {
  __shared__ _Float16 pst[8][16 * 32];  // per-wave private P staging (D->A layout)
  int wave = threadIdx.x >> 5, lane = threadIdx.x & 31;
  int b = blockIdx.x >> 3, qc = blockIdx.x & 7;
  int q0 = qc * 128 + wave * 16;
  int rowbase = b * S_ + q0;
  int rh = (lane >> 4) ? 8 : 0;
  float Lq[8];
#pragma unroll
  for (int i = 0; i < 8; ++i) Lq[i] = Lv[rowbase + rh + i];
  v8f zero = {};
  v8f num[8];
#pragma unroll
  for (int i = 0; i < 8; ++i) num[i] = zero;
  float den[8] = {0, 0, 0, 0, 0, 0, 0, 0};
  const _Float16* qb = q16 + (size_t)rowbase * 128;
  _Float16* myp = &pst[wave][0];
  int nkc = (q0 + 16 + 31) >> 5;  // causal: keys up to this wave's last row
  for (int kc = 0; kc < nkc; ++kc) {
    int s0 = kc * 32;
    // prefetch next key chunk (k tiles + transposed v tiles) into cache while
    // this chunk's WMMAs execute -> global_prefetch_b8
    if (kc + 1 < nkc) {
      int sn = s0 + 32;
      const char* pk = (const char*)(k16 + (size_t)(b * S_ + sn) * 128);
      __builtin_prefetch(pk + lane * 256, 0, 1);  // 32 lanes x 256B = 8KB tile
      const char* pv = (const char*)(ivT + (size_t)(b * 128 + lane * 4) * 1024 + sn);
      __builtin_prefetch(pv, 0, 1);
    }
    // scores P = Q @ Kn^T over feature dim (two 16-key tiles)
    v8f sc0 = zero, sc1 = zero;
#pragma unroll
    for (int kk = 0; kk < 128; kk += 32) {
      v16h a  = load_frag(qb, 128, kk);
      v16h b0 = load_frag(k16 + (size_t)(b * S_ + s0) * 128, 128, kk);
      v16h b1 = load_frag(k16 + (size_t)(b * S_ + s0 + 16) * 128, 128, kk);
      sc0 = wmma_f16(a, b0, sc0);
      sc1 = wmma_f16(a, b1, sc1);
    }
    // decay + causal mask + den accumulation + stage P*i_c as f16 A source
#pragma unroll
    for (int t2 = 0; t2 < 2; ++t2) {
      int s = s0 + t2 * 16 + (lane & 15);
      float Ls = Lv[b * S_ + s];
      float igs = igv[b * S_ + s];
      float ics = icv[b * S_ + s];
#pragma unroll
      for (int i = 0; i < 8; ++i) {
        int trow = q0 + rh + i;
        float p = (t2 ? sc1[i] : sc0[i]) * __expf(Lq[i] - Ls);
        p = (s <= trow) ? p : 0.f;
        den[i] += p * igs;  // row-sum -> q.n_t later via half-wave reduce
        myp[(rh + i) * 32 + t2 * 16 + (lane & 15)] = (_Float16)(p * ics);
      }
    }
    __builtin_amdgcn_wave_barrier();
    asm volatile("s_wait_dscnt 0" ::: "memory");
    // num += P @ Vn  (B fragments contiguous thanks to transposed ivT)
    v16h ap = load_frag(myp, 32, 0);
#pragma unroll
    for (int dt = 0; dt < 8; ++dt) {
      v16h bp = load_frag(ivT + ((size_t)(b * 128 + dt * 16)) * 1024 + s0, 1024, 0);
      num[dt] = wmma_f16(ap, bp, num[dt]);
    }
    __builtin_amdgcn_wave_barrier();
    asm volatile("s_wait_dscnt 0" ::: "memory");
  }
  // reduce den across the 16-lane half that holds this row's columns
  float rden[8];
#pragma unroll
  for (int i = 0; i < 8; ++i) {
    float d = den[i];
    d += __shfl_xor(d, 1);
    d += __shfl_xor(d, 2);
    d += __shfl_xor(d, 4);
    d += __shfl_xor(d, 8);
    rden[i] = fastRcp(fmaxf(fabsf(d), 1.0f));  // THRESH
  }
#pragma unroll
  for (int dt = 0; dt < 8; ++dt)
#pragma unroll
    for (int i = 0; i < 8; ++i)
      h32[(size_t)(rowbase + rh + i) * 128 + dt * 16 + (lane & 15)] = num[dt][i] * rden[i];
}

// o * LN(h)
__global__ void k_memout(const float* __restrict__ h32, const float* __restrict__ o32,
                         const float* __restrict__ g, const float* __restrict__ bt,
                         float* __restrict__ y1) {
  __shared__ float sh[8];
  int row = blockIdx.x, t = threadIdx.x;
  float v = h32[(size_t)row * 128 + t];
  float mu = blockSum(v, sh) * (1.f / 128.f);
  float dv = v - mu;
  float var = blockSum(dv * dv, sh) * (1.f / 128.f);
  float xn = dv * rsqrtf(var + 1e-5f) * g[t] + bt[t];
  y1[(size_t)row * 128 + t] = o32[(size_t)row * 128 + t] * xn;
}

// rolling std (w=10) of feature-mean, mean-normalized
__global__ void k_vol(const float* __restrict__ mv, float* __restrict__ vol) {
  __shared__ float sh[8];
  int b = blockIdx.x, t = threadIdx.x;
  float sv[4];
  float part = 0.f;
#pragma unroll
  for (int r = 0; r < 4; ++r) {
    int s = t + r * 256;
    int se = s < 9 ? 9 : s;  // warmup backfill from s=9
    float s1 = 0.f, s2 = 0.f;
    for (int j = 0; j < 10; ++j) {
      float m = mv[b * S_ + se - j];
      s1 += m; s2 += m * m;
    }
    float var = (s2 - s1 * s1 * 0.1f) * (1.f / 9.f);
    sv[r] = sqrtf(fmaxf(var, 1e-12f));
    part += sv[r];
  }
  float mean = blockSum(part, sh) * (1.f / 1024.f);
  float rmean = fastRcp(mean + 1e-6f);
#pragma unroll
  for (int r = 0; r < 4; ++r) vol[b * S_ + t + r * 256] = sv[r] * rmean;
}

// diagonal SSM scan (bilinear), per-batch block, one thread per d
__global__ void k_ssm(const float* __restrict__ y1, const float* __restrict__ vol,
                      const float* __restrict__ llr, const float* __restrict__ logb,
                      const float* __restrict__ cvec, const float* __restrict__ logd,
                      const float* __restrict__ logstep, const float* __restrict__ vgate,
                      const float* __restrict__ alphap,
                      const float* __restrict__ lng, const float* __restrict__ lnb,
                      _Float16* __restrict__ y16) {
  __shared__ float sh[8];
  int b = blockIdx.x, d = threadIdx.x;
  float lam = -__expf(llr[d]);
  float bv = __expf(logb[d]);
  float step = __expf(logstep[0]);
  float a_d = (2.f + step * lam) * fastRcp(2.f - step * lam);
  float b_d = step * (1.f + a_d) * bv * 0.5f;
  float dpar = __expf(logd[0]);
  float gate = fastSigmoid(vgate[d]);
  float alpha = fastSigmoid(alphap[0]);
  float cd = cvec[d], gd = lng[d], bd = lnb[d];
  float h = 0.f;
  for (int t = 0; t < S_; ++t) {
    float xt = y1[((size_t)(b * S_ + t)) * 128 + d];
    float mx = blockSum(xt, sh) * (1.f / 128.f);
    h = a_d * h + b_d * mx;
    float hm = blockSum(h, sh) * (1.f / 128.f);
    float dh = h - hm;
    float hv = blockSum(dh * dh, sh) * (1.f / 128.f);
    h = dh * rsqrtf(hv + 1e-5f) * gd + bd;
    h = h * fastRcp(1.f + gate * vol[b * S_ + t]);
    float hc = blockSum(h * cd, sh);
    float y = hc + dpar * xt;
    float o = alpha * xt + (1.f - alpha) * y;
    y16[((size_t)(b * S_ + t)) * 128 + d] = (_Float16)o;
  }
}

// ---------------- host launch ----------------
extern "C" void kernel_launch(void* const* d_in, const int* in_sizes, int n_in,
                              void* d_out, int out_size, void* d_ws, size_t ws_size,
                              hipStream_t stream) {
  (void)in_sizes; (void)n_in; (void)out_size; (void)ws_size;
  const float* x      = (const float*)d_in[0];
  const float* m_init = (const float*)d_in[3];
  const float* wq = (const float*)d_in[4];  const float* bq = (const float*)d_in[5];
  const float* wk = (const float*)d_in[6];  const float* bk = (const float*)d_in[7];
  const float* wv = (const float*)d_in[8];  const float* bvv = (const float*)d_in[9];
  const float* wi = (const float*)d_in[10]; const float* bi = (const float*)d_in[11];
  const float* wf = (const float*)d_in[12]; const float* bf = (const float*)d_in[13];
  const float* wo = (const float*)d_in[14]; const float* bo = (const float*)d_in[15];
  const float* ling = (const float*)d_in[16]; const float* linb = (const float*)d_in[17];
  const float* lmg  = (const float*)d_in[18]; const float* lmb  = (const float*)d_in[19];
  const float* wr = (const float*)d_in[20]; const float* br = (const float*)d_in[21];
  const float* rlg = (const float*)d_in[22]; const float* rlb = (const float*)d_in[23];
  const float* w2 = (const float*)d_in[24]; const float* b2 = (const float*)d_in[25];
  const float *pw[4], *pb[4], *pg[4], *pbn[4];
  for (int p = 0; p < 4; ++p) {
    pw[p]  = (const float*)d_in[26 + p * 4 + 0];
    pb[p]  = (const float*)d_in[26 + p * 4 + 1];
    pg[p]  = (const float*)d_in[26 + p * 4 + 2];
    pbn[p] = (const float*)d_in[26 + p * 4 + 3];
  }
  const float* llr     = (const float*)d_in[42];
  const float* logb    = (const float*)d_in[43];
  const float* cvec    = (const float*)d_in[44];
  const float* logd    = (const float*)d_in[45];
  const float* logstep = (const float*)d_in[46];
  const float* vgate   = (const float*)d_in[47];
  const float* alphap  = (const float*)d_in[48];
  const float* slng    = (const float*)d_in[49];
  const float* slnb    = (const float*)d_in[50];
  float* out = (float*)d_out;

  char* wsb = (char*)d_ws;
  size_t off = 0;
  auto alloc = [&](size_t bytes) -> void* {
    off = (off + 255) & ~(size_t)255;
    void* p = (void*)(wsb + off);
    off += bytes;
    return p;
  };
  _Float16* xn16 = (_Float16*)alloc((size_t)ROWS * 128 * 2);
  _Float16* wq16 = (_Float16*)alloc(16384 * 2);
  _Float16* wk16 = (_Float16*)alloc(16384 * 2);
  _Float16* wv16 = (_Float16*)alloc(16384 * 2);
  _Float16* wo16 = (_Float16*)alloc(16384 * 2);
  _Float16* wr16 = (_Float16*)alloc(16384 * 2);
  _Float16* wp16[4];
  for (int p = 0; p < 4; ++p) wp16[p] = (_Float16*)alloc(16384 * 2);
  float* q32 = (float*)alloc((size_t)ROWS * 128 * 4);
  float* k32 = (float*)alloc((size_t)ROWS * 128 * 4);
  float* v32 = (float*)alloc((size_t)ROWS * 128 * 4);
  float* o32 = (float*)alloc((size_t)ROWS * 128 * 4);
  _Float16* r16 = (_Float16*)alloc((size_t)ROWS * 128 * 2);
  float* rwv = (float*)alloc((size_t)ROWS * 4 * 4);
  float* mv  = (float*)alloc((size_t)ROWS * 4);
  float* igv = (float*)alloc((size_t)ROWS * 4);
  float* icv = (float*)alloc((size_t)ROWS * 4);
  float* fzv = (float*)alloc((size_t)ROWS * 4);
  float* Lvv = (float*)alloc((size_t)ROWS * 4);
  float* volv = (float*)alloc((size_t)ROWS * 4);
  _Float16* q16 = (_Float16*)alloc((size_t)ROWS * 128 * 2);
  _Float16* k16 = (_Float16*)alloc((size_t)ROWS * 128 * 2);
  _Float16* ivT = (_Float16*)alloc((size_t)ROWS * 128 * 2);
  float* h32 = (float*)alloc((size_t)ROWS * 128 * 4);
  float* y1  = (float*)alloc((size_t)ROWS * 128 * 4);
  _Float16* y16 = (_Float16*)alloc((size_t)ROWS * 128 * 2);

  // weight f32 -> f16
  k_f2h<<<64, 256, 0, stream>>>(wq, wq16, 16384);
  k_f2h<<<64, 256, 0, stream>>>(wk, wk16, 16384);
  k_f2h<<<64, 256, 0, stream>>>(wv, wv16, 16384);
  k_f2h<<<64, 256, 0, stream>>>(wo, wo16, 16384);
  k_f2h<<<64, 256, 0, stream>>>(wr, wr16, 16384);
  for (int p = 0; p < 4; ++p) k_f2h<<<64, 256, 0, stream>>>(pw[p], wp16[p], 16384);

  k_ln_gates<<<ROWS, 128, 0, stream>>>(x, ling, linb, wi, bi, wf, bf, m_init,
                                       xn16, mv, igv, icv, fzv);
  k_lscan<<<B_, 32, 0, stream>>>(fzv, Lvv);

  k_gemm<<<ROWS / 16, 256, 0, stream>>>(xn16, wq16, bq, q32, nullptr,
                                        nullptr, nullptr, nullptr, nullptr, 0, 0);
  k_gemm<<<ROWS / 16, 256, 0, stream>>>(xn16, wk16, bk, k32, nullptr,
                                        nullptr, nullptr, nullptr, nullptr, 0, 0);
  k_gemm<<<ROWS / 16, 256, 0, stream>>>(xn16, wv16, bvv, v32, nullptr,
                                        nullptr, nullptr, nullptr, nullptr, 0, 0);
  k_gemm<<<ROWS / 16, 256, 0, stream>>>(xn16, wo16, bo, o32, nullptr,
                                        nullptr, nullptr, nullptr, nullptr, 1, 0);
  k_gemm<<<ROWS / 16, 256, 0, stream>>>(xn16, wr16, br, nullptr, r16,
                                        rlg, rlb, nullptr, nullptr, 2, 0);

  k_regime<<<ROWS, 128, 0, stream>>>(r16, w2, b2, rwv);
  k_prep<<<ROWS, 128, 0, stream>>>(q32, k32, v32, q16, k16, ivT);
  k_attn<<<B_ * 8, 256, 0, stream>>>(q16, k16, ivT, Lvv, igv, icv, h32);
  k_memout<<<ROWS, 128, 0, stream>>>(h32, o32, lmg, lmb, y1);
  k_vol<<<B_, 256, 0, stream>>>(mv, volv);
  k_ssm<<<B_, 128, 0, stream>>>(y1, volv, llr, logb, cvec, logd, logstep,
                                vgate, alphap, slng, slnb, y16);

  for (int p = 0; p < 4; ++p)
    k_gemm<<<ROWS / 16, 256, 0, stream>>>(y16, wp16[p], pb[p], nullptr, nullptr,
                                          pg[p], pbn[p], rwv, out, 3, p);
}
